// TemporalMaliciousAgentDetector_55405078119386
// MI455X (gfx1250) — matrix operand order repaired and checked by reading, not compiled, not run on hardware
//
#include <hip/hip_runtime.h>
#include <math.h>

// ---------------- problem constants (match reference) ----------------
#define B_    2
#define T_    12
#define N_    2500
#define FIN   64
#define HID   128
#define E_    40000
#define KC    3
#define HEADS 8
#define LSTMH 64
#define BN    (B_ * N_)

typedef __attribute__((ext_vector_type(16))) __bf16 bf16x16;
typedef __attribute__((ext_vector_type(8)))  float  f32x8;

enum { ACT_NONE = 0, ACT_SIGMOID = 1, ACT_TANH = 2, ACT_RELU = 3 };

__device__ __forceinline__ float act_apply(float v, int act) {
  switch (act) {
    case ACT_SIGMOID: return 1.f / (1.f + __expf(-v));
    case ACT_TANH:    return tanhf(v);
    case ACT_RELU:    return fmaxf(v, 0.f);
    default:          return v;
  }
}

// ---------------------------------------------------------------------
// WMMA GEMM: C[M,N] = act( A[M,K] @ B[K,N] + bias (+ C if accum) )
// A row-major (stride lda, 32B-aligned rows), B row-major (ldb = N),
// C stride ldc.  CONTRACT: K % 64 == 0, lda % 8 == 0.
// One 16x16 output tile per wave; 4 waves (128 thr) per block share a
// 64x16 B tile through LDS stored FRAGMENT-MAJOR: the 16 bf16 values
// lane L needs for WMMA #sub live contiguously at Bs[(sub*32+L)*16]
// (lane = hi*16 + ml exactly matches the ISA 7.12.2 B-fragment rows),
// so each fragment is one 32B vector LDS load (2x ds_load_b128).
// Out-of-range rows/cols are CLAMPED on load (row m of A only feeds
// row m of D; col n of B only feeds col n of D; stores are guarded),
// so the hot loop has zero branches.
// ---------------------------------------------------------------------
__global__ __launch_bounds__(128) void gemm_wmma_kernel(
    const float* __restrict__ A, int lda,
    const float* __restrict__ Bw,
    const float* __restrict__ bias,
    float* __restrict__ C, int ldc,
    int M, int N, int K, int act, int accum)
{
  __shared__ __align__(32) __bf16 Bs[64 * 16];   // [sub*32 + lane][i]

  const int tid   = threadIdx.x;
  const int lane  = tid & 31;
  const int wave  = tid >> 5;
  const int tileN = blockIdx.x * 16;
  const int tileM = (blockIdx.y * 4 + wave) * 16;

  const int ml = lane & 15;
  const int hi = lane >> 4;                 // 0 or 1 (half-wave)
  const int mrow   = tileM + ml;
  const int mclamp = (mrow < M) ? mrow : (M - 1);
  const float* Arow = A + (size_t)mclamp * lda;

  // B staging coords for this thread (8 K-rows apart, unconditional)
  const int snl = tid & 15;
  const int skl = tid >> 4;                 // 0..7
  const int nst = tileN + snl;
  const int nclamp = (nst < N) ? nst : (N - 1);

  f32x8 acc = {0.f, 0.f, 0.f, 0.f, 0.f, 0.f, 0.f, 0.f};

  for (int k0 = 0; k0 < K; k0 += 64) {
    // stage B tile (64 x 16) into LDS (f32 -> bf16), fragment-major:
    // element (kl, nl) -> Bs[((kl>>5)*32 + ((kl>>3)&1)*16 + nl)*16 + (kl&7) + ((kl>>4)&1)*8]
#pragma unroll
    for (int j = 0; j < 8; ++j) {
      int kl = skl + j * 8;
      float v = Bw[(size_t)(k0 + kl) * N + nclamp];
      int slot = (kl & 32) + ((kl & 8) ? 16 : 0) + snl;
      Bs[slot * 16 + (kl & 7) + ((kl & 16) ? 8 : 0)] = (__bf16)v;
    }
    if (k0 + 64 < K)
      __builtin_prefetch(&Bw[(size_t)(k0 + 64 + skl * 8) * N + nclamp], 0, 0);
    __syncthreads();

#pragma unroll
    for (int sub = 0; sub < 2; ++sub) {
      const int ka = k0 + sub * 32 + hi * 8;
      // A fragment: lanes 0-15 row m, K = ka+0..7 & ka+16..23; lanes 16-31: +8
      float4 a0 = *(const float4*)(Arow + ka);
      float4 a1 = *(const float4*)(Arow + ka + 4);
      float4 a2 = *(const float4*)(Arow + ka + 16);
      float4 a3 = *(const float4*)(Arow + ka + 20);
      bf16x16 afrag;
      afrag[0]  = (__bf16)a0.x; afrag[1]  = (__bf16)a0.y;
      afrag[2]  = (__bf16)a0.z; afrag[3]  = (__bf16)a0.w;
      afrag[4]  = (__bf16)a1.x; afrag[5]  = (__bf16)a1.y;
      afrag[6]  = (__bf16)a1.z; afrag[7]  = (__bf16)a1.w;
      afrag[8]  = (__bf16)a2.x; afrag[9]  = (__bf16)a2.y;
      afrag[10] = (__bf16)a2.z; afrag[11] = (__bf16)a2.w;
      afrag[12] = (__bf16)a3.x; afrag[13] = (__bf16)a3.y;
      afrag[14] = (__bf16)a3.z; afrag[15] = (__bf16)a3.w;
      // B fragment: one contiguous 32B LDS vector per lane
      bf16x16 bfrag = *(const bf16x16*)&Bs[(sub * 32 + lane) * 16];

      acc = __builtin_amdgcn_wmma_f32_16x16x32_bf16(
          false, afrag, false, bfrag, (short)0, acc, false, false);
    }
    __syncthreads();
  }

  // C/D layout: VGPR r <-> M = tileM + r + 8*hi, N = tileN + (lane&15)
  const int ncol = tileN + ml;
  if (ncol < N) {
    const float bv = bias ? bias[ncol] : 0.f;
#pragma unroll
    for (int r = 0; r < 8; ++r) {
      int mr = tileM + r + hi * 8;
      if (mr < M) {
        float v = acc[r] + bv;
        if (accum) v += C[(size_t)mr * ldc + ncol];
        C[(size_t)mr * ldc + ncol] = act_apply(v, act);
      }
    }
  }
}

// ---------------- elementwise / graph kernels ----------------
__global__ void zero_kernel(float* p, int n) {
  int i = blockIdx.x * blockDim.x + threadIdx.x;
  if (i < n) p[i] = 0.f;
}

__global__ void deg_kernel(const int* __restrict__ src, const int* __restrict__ dst,
                           float* __restrict__ deg) {
  int e = blockIdx.x * blockDim.x + threadIdx.x;
  if (e >= E_) return;
  atomicAdd(&deg[src[e]], 1.f);
  atomicAdd(&deg[N_ + dst[e]], 1.f);
}

__global__ void inv_kernel(const float* __restrict__ deg, float* __restrict__ inv) {
  int i = blockIdx.x * blockDim.x + threadIdx.x;
  if (i >= 2 * N_) return;
  float d = deg[i];
  inv[i] = (d > 0.f) ? 1.f / fmaxf(d, 1.f) : 0.f;
}

__global__ void load_x_kernel(const float* __restrict__ x, float* __restrict__ h, int t) {
  int idx = blockIdx.x * blockDim.x + threadIdx.x;
  if (idx >= B_ * N_ * FIN) return;
  int f = idx % FIN;
  int n = (idx / FIN) % N_;
  int b = idx / (FIN * N_);
  h[idx] = x[(((size_t)b * T_ + t) * N_ + n) * FIN + f];
}

// XC row (b*N+n) of width 5*Din: block0 = h, blocks 1..4 zeroed
__global__ void xc_init_kernel(const float* __restrict__ h, float* __restrict__ XC, int Din) {
  int ld5 = 5 * Din;
  int total = BN * ld5;
  int idx = blockIdx.x * blockDim.x + threadIdx.x;
  if (idx >= total) return;
  int col = idx % ld5;
  int row = idx / ld5;
  XC[idx] = (col < Din) ? h[(size_t)row * Din + col] : 0.f;
}

// out[b, to[e], colO+f] += XC[b, from[e], colX+f] * w[from[e]]
__global__ void scatter_prop_kernel(float* __restrict__ XC,
                                    const int* __restrict__ toIdx,
                                    const int* __restrict__ fromIdx,
                                    const float* __restrict__ w,
                                    int Din, int ld5, int colX, int colO, int total) {
  int idx = blockIdx.x * blockDim.x + threadIdx.x;
  if (idx >= total) return;
  int f = idx % Din;
  int e = (idx / Din) % E_;
  int b = idx / (Din * E_);
  int s = fromIdx[e];
  int d = toIdx[e];
  float val = XC[((size_t)(b * N_ + s)) * ld5 + colX + f] * w[s];
  atomicAdd(&XC[((size_t)(b * N_ + d)) * ld5 + colO + f], val);
}

// Chebyshev recurrence finalize: X2 = 2*P(X1) - X0 for both directions
__global__ void cheb_fin_kernel(float* __restrict__ XC, int Din) {
  int idx = blockIdx.x * blockDim.x + threadIdx.x;
  if (idx >= BN * Din) return;
  int f = idx % Din;
  int row = idx / Din;
  int ld5 = 5 * Din;
  size_t base = (size_t)row * ld5;
  float t0 = XC[base + f];
  XC[base + 3 * Din + f] = 2.f * XC[base + 3 * Din + f] - t0;
  XC[base + 4 * Din + f] = 2.f * XC[base + 4 * Din + f] - t0;
}

// h_next = (1 - Z) * tanh(Hc)   (Z already sigmoid'd, H already tanh'd)
__global__ void gru_update_kernel(const float* __restrict__ Z, const float* __restrict__ H,
                                  float* __restrict__ hn) {
  int idx = blockIdx.x * blockDim.x + threadIdx.x;
  if (idx >= BN * HID) return;
  hn[idx] = (1.f - Z[idx]) * H[idx];
}

__global__ void store_temporal_kernel(const float* __restrict__ h, float* __restrict__ tmp, int t) {
  int idx = blockIdx.x * blockDim.x + threadIdx.x;
  if (idx >= BN * HID) return;
  int d = idx % HID;
  int n = (idx / HID) % N_;
  int b = idx / (HID * N_);
  tmp[(((size_t)b * T_ + t) * N_ + n) * HID + d] = h[idx];
}

// (B,T,N,H) -> (B*N, T, H)
__global__ void build_attn_in_kernel(const float* __restrict__ tmp, float* __restrict__ ai) {
  int idx = blockIdx.x * blockDim.x + threadIdx.x;
  if (idx >= B_ * N_ * T_ * HID) return;
  int d = idx % HID;
  int t = (idx / HID) % T_;
  int n = (idx / (HID * T_)) % N_;
  int b = idx / (HID * T_ * N_);
  ai[idx] = tmp[(((size_t)b * T_ + t) * N_ + n) * HID + d];
}

// per (bn, head, t): softmax(q.k/4) @ v ; dh = 16, T = 12
__global__ void mha_attn_kernel(const float* __restrict__ qkv, float* __restrict__ o) {
  int idx = blockIdx.x * blockDim.x + threadIdx.x;
  if (idx >= BN * HEADS * T_) return;
  int t  = idx % T_;
  int hh = (idx / T_) % HEADS;
  int bn = idx / (T_ * HEADS);
  const float* base = qkv + (size_t)bn * T_ * (3 * HID);
  const float* qrow = base + (size_t)t * (3 * HID) + hh * 16;
  float qv[16];
#pragma unroll
  for (int d = 0; d < 16; ++d) qv[d] = qrow[d];
  float s[T_];
  float mx = -3.402823466e38f;
  for (int u = 0; u < T_; ++u) {
    const float* krow = base + (size_t)u * (3 * HID) + HID + hh * 16;
    float a = 0.f;
#pragma unroll
    for (int d = 0; d < 16; ++d) a += qv[d] * krow[d];
    a *= 0.25f;           // 1/sqrt(dh)
    s[u] = a;
    mx = fmaxf(mx, a);
  }
  float den = 0.f;
  for (int u = 0; u < T_; ++u) { s[u] = __expf(s[u] - mx); den += s[u]; }
  float inv = 1.f / den;
  float ov[16];
#pragma unroll
  for (int d = 0; d < 16; ++d) ov[d] = 0.f;
  for (int u = 0; u < T_; ++u) {
    const float* vrow = base + (size_t)u * (3 * HID) + 2 * HID + hh * 16;
    float a = s[u] * inv;
#pragma unroll
    for (int d = 0; d < 16; ++d) ov[d] += a * vrow[d];
  }
  float* orow = o + (size_t)bn * T_ * HID + (size_t)t * HID + hh * 16;
#pragma unroll
  for (int d = 0; d < 16; ++d) orow[d] = ov[d];
}

// gates (BN,256) split i|f|g|o ; update h,c ; write h into out[bn,t,:64] (pre-offset for dir)
__global__ void lstm_cell_kernel(const float* __restrict__ gates,
                                 float* __restrict__ h, float* __restrict__ c,
                                 float* __restrict__ outbase, int t) {
  int idx = blockIdx.x * blockDim.x + threadIdx.x;
  if (idx >= BN * LSTMH) return;
  int j  = idx % LSTMH;
  int bn = idx / LSTMH;
  const float* g = gates + (size_t)bn * 4 * LSTMH;
  float gi = 1.f / (1.f + __expf(-g[j]));
  float gf = 1.f / (1.f + __expf(-g[LSTMH + j]));
  float gg = tanhf(g[2 * LSTMH + j]);
  float go = 1.f / (1.f + __expf(-g[3 * LSTMH + j]));
  float cn = gf * c[idx] + gi * gg;
  c[idx] = cn;
  float hn = go * tanhf(cn);
  h[idx] = hn;
  outbase[(size_t)bn * T_ * HID + (size_t)t * HID + j] = hn;
}

// grep[:,0:128] = mean_T, grep[:,128:256] = max_T
__global__ void pool_kernel(const float* __restrict__ lo, float* __restrict__ grep) {
  int idx = blockIdx.x * blockDim.x + threadIdx.x;
  if (idx >= BN * HID) return;
  int d  = idx % HID;
  int bn = idx / HID;
  float s = 0.f, m = -3.402823466e38f;
  for (int t = 0; t < T_; ++t) {
    float v = lo[((size_t)bn * T_ + t) * HID + d];
    s += v;
    m = fmaxf(m, v);
  }
  grep[(size_t)bn * (2 * HID) + d] = s / (float)T_;
  grep[(size_t)bn * (2 * HID) + HID + d] = m;
}

// Pack DCRNN gate weight (2,K,cin,HID) -> (5*Din, HID):
// [W(0,0)+W(1,0) ; W(0,1) ; W(1,1) ; W(0,2) ; W(1,2)]  (rows 0..Din of cin axis;
// the remaining cin rows multiply the all-zero h0 half and are dropped)
__global__ void pack_w_kernel(const float* __restrict__ W, float* __restrict__ Wp,
                              int cin, int Din) {
  int total = 5 * Din * HID;
  int idx = blockIdx.x * blockDim.x + threadIdx.x;
  if (idx >= total) return;
  int hcol = idx % HID;
  int row  = idx / HID;
  int blk  = row / Din;
  int f    = row % Din;
  float v;
  if (blk == 0) {
    v = W[((size_t)(0 * KC + 0) * cin + f) * HID + hcol] +
        W[((size_t)(1 * KC + 0) * cin + f) * HID + hcol];
  } else {
    int dd = (blk == 1 || blk == 3) ? 0 : 1;
    int kk = (blk <= 2) ? 1 : 2;
    v = W[((size_t)(dd * KC + kk) * cin + f) * HID + hcol];
  }
  Wp[idx] = v;
}

#define L1(KERN, n, ...) KERN<<<dim3((unsigned)(((n) + 255) / 256)), dim3(256), 0, stream>>>(__VA_ARGS__)

extern "C" void kernel_launch(void* const* d_in, const int* in_sizes, int n_in,
                              void* d_out_v, int out_size, void* d_ws, size_t ws_size,
                              hipStream_t stream) {
  (void)in_sizes; (void)n_in; (void)out_size; (void)ws_size;

  // ---- input mapping (top-level dict order, params pytree: sorted keys) ----
  const float* x  = (const float*)d_in[0];        // (B,T,N,FIN)
  const int*   ei = (const int*)  d_in[1];        // (T,2,E)
  // d_in[2] edge_attr unused (reference deletes it)
  const float* anom_W1 = (const float*)d_in[3];   // (128,64)
  const float* anom_W2 = (const float*)d_in[4];   // (64,1)
  const float* anom_b1 = (const float*)d_in[5];
  const float* anom_b2 = (const float*)d_in[6];
  const float* attn_Wo   = (const float*)d_in[7];   // (128,128)
  const float* attn_Wqkv = (const float*)d_in[8];   // (128,384)
  const float* attn_bo   = (const float*)d_in[9];
  const float* attn_bqkv = (const float*)d_in[10];
  const float* cls_W1 = (const float*)d_in[11];     // (256,128)
  const float* cls_W2 = (const float*)d_in[12];     // (128,2)
  const float* cls_b1 = (const float*)d_in[13];
  const float* cls_b2 = (const float*)d_in[14];
  // dcrnn: 3 layers, each sorted {Wh, Wr, Wz, bh, br, bz}; R gate provably unused (h0==0)
  const float *dc_Wh[3], *dc_Wz[3], *dc_bh[3], *dc_bz[3];
  for (int l = 0; l < 3; ++l) {
    int base = 15 + l * 6;
    dc_Wh[l] = (const float*)d_in[base + 0];
    dc_Wz[l] = (const float*)d_in[base + 2];
    dc_bh[l] = (const float*)d_in[base + 3];
    dc_bz[l] = (const float*)d_in[base + 5];
  }
  // lstm: 2 layers, each sorted {bwd, fwd}, each dir sorted {Whh, Wih, bhh, bih}
  const float *ls_Whh[2][2], *ls_Wih[2][2], *ls_bhh[2][2], *ls_bih[2][2];
  for (int l = 0; l < 2; ++l)
    for (int d = 0; d < 2; ++d) {          // d: 0 = bwd, 1 = fwd (sorted key order)
      int base = 33 + l * 8 + d * 4;
      ls_Whh[l][d] = (const float*)d_in[base + 0];
      ls_Wih[l][d] = (const float*)d_in[base + 1];
      ls_bhh[l][d] = (const float*)d_in[base + 2];
      ls_bih[l][d] = (const float*)d_in[base + 3];
    }

  float* out      = (float*)d_out_v;
  float* logits   = out;             // (5000, 2)
  float* anom_s   = out + 10000;     // (5000,)
  float* temporal = out + 15000;     // (B,T,N,HID) -- third output, written in place

  // ---- workspace layout (floats): R0/R1 = 7.68M each, R2 = 23.04M ----
  float* ws = (float*)d_ws;
  float* R0 = ws;                    // attn_in -> mha_out -> lstm ping
  float* R1 = ws + 7680000;          // attn o  -> lstm pong
  float* R2 = ws + 15360000;         // qkv; time-shared DCRNN / LSTM scratch
  // DCRNN-phase scratch inside R2
  float* XC = R2;                    // (BN, 5*Din)  max 3.2M
  float* hA = R2 + 3200000;          // (BN, 128)
  float* hB = R2 + 3840000;
  float* Zb = R2 + 4480000;
  float* Hb = R2 + 5120000;
  float* packs = R2 + 5760000;       // packed gate weights (409600)
  float* wzp[3], *whp[3];
  {
    size_t po = 0;
    for (int l = 0; l < 3; ++l) {
      int Din = l ? HID : FIN;
      wzp[l] = packs + po; po += (size_t)5 * Din * HID;
      whp[l] = packs + po; po += (size_t)5 * Din * HID;
    }
  }
  float* deg = R2 + 6200000;         // deg_out[N] | deg_in[N]
  float* inv = deg + 2 * N_;         // inv_out[N] | inv_in[N]

  auto gemm = [&](const float* A, int lda, const float* Bw, const float* bias,
                  float* C, int ldc, int M, int Nn, int Kk, int act, int accum) {
    dim3 grid((unsigned)((Nn + 15) / 16), (unsigned)((M + 63) / 64));
    gemm_wmma_kernel<<<grid, dim3(128), 0, stream>>>(A, lda, Bw, bias, C, ldc,
                                                     M, Nn, Kk, act, accum);
  };

  // ---- pack DCRNN gate weights once ----
  for (int l = 0; l < 3; ++l) {
    int cin = (l ? HID : FIN) + HID;
    int Din = l ? HID : FIN;
    int n5 = 5 * Din * HID;
    L1(pack_w_kernel, n5, dc_Wz[l], wzp[l], cin, Din);
    L1(pack_w_kernel, n5, dc_Wh[l], whp[l], cin, Din);
  }

  // ---- DCRNN encoder, timestep loop ----
  for (int t = 0; t < T_; ++t) {
    const int* src = ei + (size_t)t * 2 * E_;
    const int* dst = src + E_;
    L1(zero_kernel, 2 * N_, deg, 2 * N_);
    L1(deg_kernel, E_, src, dst, deg);
    L1(inv_kernel, 2 * N_, deg, inv);
    L1(load_x_kernel, B_ * N_ * FIN, x, hA, t);
    float* hcur = hA;
    float* hnxt = hB;
    for (int l = 0; l < 3; ++l) {
      int Din = l ? HID : FIN;
      int ld5 = 5 * Din;
      L1(xc_init_kernel, BN * ld5, hcur, XC, Din);
      int npe = B_ * E_ * Din;
      // T1_fwd = pf(h), T1_bwd = pb(h)
      L1(scatter_prop_kernel, npe, XC, dst, src, inv,       Din, ld5, 0,       Din,     npe);
      L1(scatter_prop_kernel, npe, XC, src, dst, inv + N_,  Din, ld5, 0,       2 * Din, npe);
      // P(T1) into T2 slots, then finalize T2 = 2*P - T0
      L1(scatter_prop_kernel, npe, XC, dst, src, inv,       Din, ld5, Din,     3 * Din, npe);
      L1(scatter_prop_kernel, npe, XC, src, dst, inv + N_,  Din, ld5, 2 * Din, 4 * Din, npe);
      L1(cheb_fin_kernel, BN * Din, XC, Din);
      gemm(XC, ld5, wzp[l], dc_bz[l], Zb, HID, BN, HID, ld5, ACT_SIGMOID, 0);
      gemm(XC, ld5, whp[l], dc_bh[l], Hb, HID, BN, HID, ld5, ACT_TANH, 0);
      L1(gru_update_kernel, BN * HID, Zb, Hb, hnxt);
      float* tp = hcur; hcur = hnxt; hnxt = tp;
    }
    L1(store_temporal_kernel, BN * HID, hcur, temporal, t);
  }

  // ---- MHA ----
  L1(build_attn_in_kernel, B_ * N_ * T_ * HID, temporal, R0);
  gemm(R0, HID, attn_Wqkv, attn_bqkv, R2, 3 * HID, BN * T_, 3 * HID, HID, ACT_NONE, 0);
  L1(mha_attn_kernel, BN * HEADS * T_, R2, R1);
  gemm(R1, HID, attn_Wo, attn_bo, R0, HID, BN * T_, HID, HID, ACT_NONE, 0);

  // ---- BiLSTM (2 layers) ----
  float* gates = R2;                 // (BN, 256)
  float* hb    = R2 + 1280000;       // (BN, 64)
  float* cb    = R2 + 1600000;       // (BN, 64)
  float* grep  = R2 + 1920000;       // (BN, 256)
  float* clsh  = R2 + 3200000;       // (BN, 128)
  float* anomh = R2 + 3840000;       // (BN, 64)
  float* xbuf = R0;
  float* obuf = R1;
  for (int l = 0; l < 2; ++l) {
    for (int dir = 0; dir < 2; ++dir) {      // 0 = forward, 1 = backward
      int pidx = (dir == 0) ? 1 : 0;         // param index: 1 = fwd, 0 = bwd
      L1(zero_kernel, BN * LSTMH, hb, BN * LSTMH);
      L1(zero_kernel, BN * LSTMH, cb, BN * LSTMH);
      for (int s = 0; s < T_; ++s) {
        int t = (dir == 0) ? s : (T_ - 1 - s);
        gemm(xbuf + (size_t)t * HID, T_ * HID, ls_Wih[l][pidx], ls_bih[l][pidx],
             gates, 4 * LSTMH, BN, 4 * LSTMH, HID, ACT_NONE, 0);
        gemm(hb, LSTMH, ls_Whh[l][pidx], ls_bhh[l][pidx],
             gates, 4 * LSTMH, BN, 4 * LSTMH, LSTMH, ACT_NONE, 1);
        L1(lstm_cell_kernel, BN * LSTMH, gates, hb, cb, obuf + dir * LSTMH, t);
      }
    }
    float* tp = xbuf; xbuf = obuf; obuf = tp;
  }
  float* lstm_out = xbuf;            // (BN, T, 128)

  // ---- heads ----
  L1(pool_kernel, BN * HID, lstm_out, grep);
  gemm(grep, 2 * HID, cls_W1, cls_b1, clsh, HID, BN, HID, 2 * HID, ACT_RELU, 0);
  gemm(clsh, HID, cls_W2, cls_b2, logits, 2, BN, 2, HID, ACT_NONE, 0);
  gemm(lstm_out + (size_t)(T_ - 1) * HID, T_ * HID, anom_W1, anom_b1,
       anomh, LSTMH, BN, LSTMH, HID, ACT_RELU, 0);
  gemm(anomh, LSTMH, anom_W2, anom_b2, anom_s, 1, BN, 1, LSTMH, ACT_SIGMOID, 0);
}
